// DeformableConv_73538430042701
// MI455X (gfx1250) — matrix operand order repaired and checked by reading, not compiled, not run on hardware
//
#include <hip/hip_runtime.h>
#include <hip/hip_bf16.h>
#include <math.h>

// ---------------------------------------------------------------------------
// Problem constants (match reference)
// ---------------------------------------------------------------------------
#define BB   8
#define HH   64
#define WW   64
#define CC   128
#define GG   8
#define PP   9            // K*K
#define HID  256
#define MM   (BB*HH*WW)   // 32768 rows for all GEMMs
#define HWP  (HH*WW)      // 4096

typedef __bf16 v16bf __attribute__((ext_vector_type(16)));
typedef __bf16 v8bf  __attribute__((ext_vector_type(8)));
typedef float  v8f   __attribute__((ext_vector_type(8)));

// ---------------------------------------------------------------------------
// Workspace layout (bytes)
// ---------------------------------------------------------------------------
#define WS_WPROJ   0u                      // 128*128 bf16       = 32768
#define WS_WOFF    32768u                  // 144*128 bf16       = 36864
#define WS_WMASK   69632u                  // 80*128 bf16 (pad)  = 20480
#define WS_WFC1    90112u                  // 256*128 bf16       = 65536
#define WS_WFC2    155648u                 // 128*256 bf16       = 65536
#define WS_STATS   221184u                 // 256 f32            = 1024
#define WS_BPROJ   222208u                 // 128 f32 (padded bias)
#define WS_BOFF    222720u                 // 144 f32
#define WS_BMASK   223296u                 // 80  f32 (pad of 72)
#define WS_BFC1    223616u                 // 256 f32
#define WS_BFC2    224640u                 // 128 f32  -> ends 225152
#define WS_INPBF   262144u                 // M*128 bf16         = 8 MB
#define WS_X1BF    8650752u                // M*128 bf16         = 8 MB
#define WS_YBUF    17039360u               // M*128 f32          = 16 MB (reused as x_proj)
#define WS_OFFSET  33816576u               // M*144 f32          = 18.87 MB (reused as hdn bf16)
#define WS_MASKP   52690944u               // M*80  f32          = 10.48 MB
#define WS_DCNBF   63176704u               // M*128 bf16         = 8 MB
// peak ~71.6 MB

__device__ __forceinline__ float gelu_exact(float x) {
    return 0.5f * x * (1.0f + erff(x * 0.70710678118654752f));
}

// ---------------------------------------------------------------------------
// f32 -> bf16 convert
// ---------------------------------------------------------------------------
__global__ __launch_bounds__(256) void f2bf_kernel(const float* __restrict__ s,
                                                   __bf16* __restrict__ d, int n) {
    int i = blockIdx.x * 256 + threadIdx.x;
    if (i < n) d[i] = (__bf16)s[i];
}

// ---------------------------------------------------------------------------
// zero-padded bias copy (removes divergent guard from GEMM epilogue)
// ---------------------------------------------------------------------------
__global__ __launch_bounds__(256) void pad_bias_kernel(const float* __restrict__ s,
                                                       float* __restrict__ d,
                                                       int nReal, int nPad) {
    int i = blockIdx.x * 256 + threadIdx.x;
    if (i < nPad) d[i] = (i < nReal) ? s[i] : 0.0f;
}

// ---------------------------------------------------------------------------
// Depthwise 3x3 conv + BN batch stats.
// Block handles 64 pixels x 128 channels; per-channel partials via LDS atomics.
// ---------------------------------------------------------------------------
__global__ __launch_bounds__(256) void dwconv_stats_kernel(
    const float* __restrict__ inp, const float* __restrict__ dw_w,
    const float* __restrict__ dw_b, float* __restrict__ y,
    float* __restrict__ stats) {
    __shared__ float lsum[CC];
    __shared__ float lsq[CC];
    if (threadIdx.x < CC) { lsum[threadIdx.x] = 0.f; lsq[threadIdx.x] = 0.f; }
    __syncthreads();

    int pix0 = blockIdx.x * 64;
    for (int it = 0; it < 32; ++it) {
        int idx = it * 256 + (int)threadIdx.x;   // 0..8191
        int c  = idx & 127;
        int pl = idx >> 7;                       // 0..63
        int pid = pix0 + pl;
        int b  = pid >> 12;
        int hw = pid & 4095;
        int h  = hw >> 6;
        int w  = hw & 63;
        float acc = dw_b[c];
        #pragma unroll
        for (int ky = 0; ky < 3; ++ky) {
            int yy = h + ky - 1;
            if (yy < 0 || yy >= HH) continue;
            #pragma unroll
            for (int kx = 0; kx < 3; ++kx) {
                int xx = w + kx - 1;
                if (xx < 0 || xx >= WW) continue;
                acc += inp[(((size_t)b * HH + yy) * WW + xx) * CC + c] *
                       dw_w[c * 9 + ky * 3 + kx];
            }
        }
        y[(size_t)pid * CC + c] = acc;
        atomicAdd(&lsum[c], acc);
        atomicAdd(&lsq[c], acc * acc);
    }
    __syncthreads();
    if (threadIdx.x < CC) atomicAdd(&stats[threadIdx.x], lsum[threadIdx.x]);
    else                  atomicAdd(&stats[threadIdx.x], lsq[threadIdx.x - CC]);
}

// ---------------------------------------------------------------------------
// BN (batch stats) + exact GELU -> bf16 x1
// ---------------------------------------------------------------------------
__global__ __launch_bounds__(256) void bn_gelu_kernel(
    const float* __restrict__ y, const float* __restrict__ stats,
    const float* __restrict__ gamma, const float* __restrict__ beta,
    __bf16* __restrict__ x1) {
    size_t i = (size_t)blockIdx.x * 256 + threadIdx.x;   // over M*C
    int c = (int)(i & 127);
    const float invN = 1.0f / (float)MM;
    float mean = stats[c] * invN;
    float var  = stats[CC + c] * invN - mean * mean;
    float inv  = rsqrtf(var + 1e-5f);
    float v = (y[i] - mean) * inv * gamma[c] + beta[c];
    x1[i] = (__bf16)gelu_exact(v);
}

// ---------------------------------------------------------------------------
// Generic bf16 WMMA GEMM:  O[M][N] = A[M][K] * Wt[N][K]^T + bias
// One wave owns a 16-row strip; all KFRAGS A-fragments stay resident in VGPRs
// (compile-time indexing, no spill) and are reused across all N tiles. Per
// N tile, ALL B fragments are preloaded first (one clause of b128 loads) so
// the WMMA chain drains the load counter progressively instead of issuing
// load->wait0->wmma lockstep. Wave32 CDNA5 fragment layout:
//   A lane l : row l%16, two 8-elem contiguous K chunks selected by l/16
//   B lane l : col l%16, two 8-elem contiguous K chunks selected by l/16
// Epilogues: 0 = f32 store, 1 = bf16 store, 2 = GELU + bf16, 3 = NCHW f32.
// ---------------------------------------------------------------------------
template <int EPI, int KFRAGS>
__global__ __launch_bounds__(256) void wmma_gemm_kernel(
    const __bf16* __restrict__ A, const __bf16* __restrict__ Wt,
    const float* __restrict__ bias, void* __restrict__ Out,
    int Mtotal, int N) {
    constexpr int Kdim = KFRAGS * 32;
    int wave = threadIdx.x >> 5;
    int lane = threadIdx.x & 31;
    int hv   = lane >> 4;          // half-wave selector
    int l16  = lane & 15;
    int mBase = (blockIdx.x * 8 + wave) * 16;
    if (mBase >= Mtotal) return;

    v16bf afr[KFRAGS];
    const __bf16* arow = A + (size_t)(mBase + l16) * Kdim;
    #pragma unroll
    for (int kf = 0; kf < KFRAGS; ++kf) {
        int kb = kf * 32;
        v8bf c0 = *(const v8bf*)(arow + kb + hv * 8);
        v8bf c1 = *(const v8bf*)(arow + kb + 16 + hv * 8);
        afr[kf] = __builtin_shufflevector(c0, c1, 0, 1, 2, 3, 4, 5, 6, 7,
                                          8, 9, 10, 11, 12, 13, 14, 15);
    }

    int nTiles = N >> 4;
    for (int nt = 0; nt < nTiles; ++nt) {
        int n = nt * 16 + l16;
        const __bf16* wrow = Wt + (size_t)n * Kdim;
        // preload all B fragments (maximizes in-wave memory parallelism)
        v16bf bfr[KFRAGS];
        #pragma unroll
        for (int kf = 0; kf < KFRAGS; ++kf) {
            v8bf d0 = *(const v8bf*)(wrow + kf * 32 + hv * 8);
            v8bf d1 = *(const v8bf*)(wrow + kf * 32 + 16 + hv * 8);
            bfr[kf] = __builtin_shufflevector(d0, d1, 0, 1, 2, 3, 4, 5, 6, 7,
                                              8, 9, 10, 11, 12, 13, 14, 15);
        }
        v8f acc = {};
        #pragma unroll
        for (int kf = 0; kf < KFRAGS; ++kf)
            acc = __builtin_amdgcn_wmma_f32_16x16x32_bf16(
                false, afr[kf], false, bfr[kf], (short)0, acc, false, false);
        float bv = bias[n];
        #pragma unroll
        for (int v = 0; v < 8; ++v) {
            int row = mBase + v + hv * 8;
            float x = acc[v] + bv;
            if (EPI == 0) {
                ((float*)Out)[(size_t)row * N + n] = x;
            } else if (EPI == 1) {
                ((__bf16*)Out)[(size_t)row * N + n] = (__bf16)x;
            } else if (EPI == 2) {
                ((__bf16*)Out)[(size_t)row * N + n] = (__bf16)gelu_exact(x);
            } else {  // NCHW: row = (b*H+h)*W+w, col n = channel
                int bimg = row >> 12;
                int hw   = row & 4095;
                ((float*)Out)[((size_t)bimg * CC + n) * HWP + hw] = x;
            }
        }
    }
}

// ---------------------------------------------------------------------------
// DCNv3 core: one thread per (pixel, group, channel-in-group).
// Local softmax over the 9 points, zero-padded bilinear gathers from the f32
// proj output. OS=1, stride=1, pad=1, dil=1 collapse the reference's
// normalized coords to: sample at (w + kx-1 + dx, h + ky-1 + dy) unpadded.
// ---------------------------------------------------------------------------
__global__ __launch_bounds__(256) void dcnv3_kernel(
    const float* __restrict__ xproj, const float* __restrict__ offs,
    const float* __restrict__ maskp, __bf16* __restrict__ out) {
    int gid = blockIdx.x * 256 + threadIdx.x;   // over M*G*16
    int c   = gid & 15;
    int g   = (gid >> 4) & 7;
    int pid = gid >> 7;
    int b   = pid >> 12;
    int hw  = pid & 4095;
    int h   = hw >> 6;
    int w   = hw & 63;

    const float* orow = offs + (size_t)pid * 144 + g * 18;
    const float* mrow = maskp + (size_t)pid * 80 + g * 9;

    float mv[PP];
    float mx = -1e30f;
    #pragma unroll
    for (int p = 0; p < PP; ++p) { mv[p] = mrow[p]; mx = fmaxf(mx, mv[p]); }
    float s = 0.f;
    #pragma unroll
    for (int p = 0; p < PP; ++p) { mv[p] = __expf(mv[p] - mx); s += mv[p]; }
    float minv = 1.0f / s;

    int ch = g * 16 + c;
    const float* xb = xproj + (size_t)b * HWP * CC + ch;
    float acc = 0.f;
    #pragma unroll
    for (int p = 0; p < PP; ++p) {
        float dx = orow[p * 2 + 0];
        float dy = orow[p * 2 + 1];
        float px = (float)w + (float)(p % 3) - 1.0f + dx;
        float py = (float)h + (float)(p / 3) - 1.0f + dy;
        float fx = floorf(px), fy = floorf(py);
        int x0 = (int)fx, y0 = (int)fy;
        float wx = px - fx, wy = py - fy;
        float v00 = 0.f, v01 = 0.f, v10 = 0.f, v11 = 0.f;
        bool xi0 = (x0 >= 0) & (x0 < WW), xi1 = (x0 + 1 >= 0) & (x0 + 1 < WW);
        bool yi0 = (y0 >= 0) & (y0 < HH), yi1 = (y0 + 1 >= 0) & (y0 + 1 < HH);
        if (yi0 & xi0) v00 = xb[((size_t)y0 * WW + x0) * CC];
        if (yi0 & xi1) v01 = xb[((size_t)y0 * WW + x0 + 1) * CC];
        if (yi1 & xi0) v10 = xb[((size_t)(y0 + 1) * WW + x0) * CC];
        if (yi1 & xi1) v11 = xb[((size_t)(y0 + 1) * WW + x0 + 1) * CC];
        float vv = v00 * (1.f - wy) * (1.f - wx) + v01 * (1.f - wy) * wx +
                   v10 * wy * (1.f - wx) + v11 * wy * wx;
        acc += vv * mv[p] * minv;
    }
    out[(size_t)pid * CC + ch] = (__bf16)acc;
}

// ---------------------------------------------------------------------------
// Launch
// ---------------------------------------------------------------------------
static inline int cdiv(int a, int b) { return (a + b - 1) / b; }

extern "C" void kernel_launch(void* const* d_in, const int* in_sizes, int n_in,
                              void* d_out, int out_size, void* d_ws, size_t ws_size,
                              hipStream_t stream) {
    const float* inp    = (const float*)d_in[0];
    const float* w_proj = (const float*)d_in[1];
    const float* b_proj = (const float*)d_in[2];
    const float* dw_w   = (const float*)d_in[3];
    const float* dw_b   = (const float*)d_in[4];
    const float* gamma  = (const float*)d_in[5];
    const float* beta   = (const float*)d_in[6];
    const float* w_off  = (const float*)d_in[7];
    const float* b_off  = (const float*)d_in[8];
    const float* w_mask = (const float*)d_in[9];
    const float* b_mask = (const float*)d_in[10];
    const float* w_fc1  = (const float*)d_in[11];
    const float* b_fc1  = (const float*)d_in[12];
    const float* w_fc2  = (const float*)d_in[13];
    const float* b_fc2  = (const float*)d_in[14];

    char* ws = (char*)d_ws;
    __bf16* wb_proj = (__bf16*)(ws + WS_WPROJ);
    __bf16* wb_off  = (__bf16*)(ws + WS_WOFF);
    __bf16* wb_mask = (__bf16*)(ws + WS_WMASK);
    __bf16* wb_fc1  = (__bf16*)(ws + WS_WFC1);
    __bf16* wb_fc2  = (__bf16*)(ws + WS_WFC2);
    float*  stats   = (float*)(ws + WS_STATS);
    float*  bp_proj = (float*)(ws + WS_BPROJ);
    float*  bp_off  = (float*)(ws + WS_BOFF);
    float*  bp_mask = (float*)(ws + WS_BMASK);
    float*  bp_fc1  = (float*)(ws + WS_BFC1);
    float*  bp_fc2  = (float*)(ws + WS_BFC2);
    __bf16* inp_bf  = (__bf16*)(ws + WS_INPBF);
    __bf16* x1_bf   = (__bf16*)(ws + WS_X1BF);
    float*  ybuf    = (float*)(ws + WS_YBUF);     // dwconv out, later x_proj
    float*  offbuf  = (float*)(ws + WS_OFFSET);   // later hdn bf16
    float*  maskbuf = (float*)(ws + WS_MASKP);
    __bf16* dcn_bf  = (__bf16*)(ws + WS_DCNBF);
    __bf16* hdn_bf  = (__bf16*)(ws + WS_OFFSET);  // reuse after dcnv3 consumed offsets

    hipMemsetAsync(stats, 0, 2 * CC * sizeof(float), stream);

    // weight + input conversions to bf16; padded bias copies
    f2bf_kernel<<<cdiv(CC * CC, 256), 256, 0, stream>>>(w_proj, wb_proj, CC * CC);
    f2bf_kernel<<<cdiv(144 * CC, 256), 256, 0, stream>>>(w_off, wb_off, 144 * CC);
    f2bf_kernel<<<cdiv(72 * CC, 256), 256, 0, stream>>>(w_mask, wb_mask, 72 * CC);
    f2bf_kernel<<<cdiv(HID * CC, 256), 256, 0, stream>>>(w_fc1, wb_fc1, HID * CC);
    f2bf_kernel<<<cdiv(CC * HID, 256), 256, 0, stream>>>(w_fc2, wb_fc2, CC * HID);
    f2bf_kernel<<<cdiv(MM * CC, 256), 256, 0, stream>>>(inp, inp_bf, MM * CC);
    pad_bias_kernel<<<1, 256, 0, stream>>>(b_proj, bp_proj, CC, CC);
    pad_bias_kernel<<<1, 256, 0, stream>>>(b_off, bp_off, 144, 144);
    pad_bias_kernel<<<1, 256, 0, stream>>>(b_mask, bp_mask, 72, 80);
    pad_bias_kernel<<<1, 256, 0, stream>>>(b_fc1, bp_fc1, HID, HID);
    pad_bias_kernel<<<1, 256, 0, stream>>>(b_fc2, bp_fc2, CC, CC);

    // depthwise conv + batch stats, then BN + GELU -> x1 (bf16)
    dwconv_stats_kernel<<<MM / 64, 256, 0, stream>>>(inp, dw_w, dw_b, ybuf, stats);
    bn_gelu_kernel<<<MM * CC / 256, 256, 0, stream>>>(ybuf, stats, gamma, beta, x1_bf);

    // input projection (f32 out, reuses ybuf): M=32768, N=128, K=128
    wmma_gemm_kernel<0, 4><<<MM / 128, 256, 0, stream>>>(
        inp_bf, wb_proj, bp_proj, ybuf, MM, CC);
    // offsets: N=144
    wmma_gemm_kernel<0, 4><<<MM / 128, 256, 0, stream>>>(
        x1_bf, wb_off, bp_off, offbuf, MM, 144);
    // mask logits: N=72 padded to 80
    wmma_gemm_kernel<0, 4><<<MM / 128, 256, 0, stream>>>(
        x1_bf, wb_mask, bp_mask, maskbuf, MM, 80);

    // deformable sampling core
    dcnv3_kernel<<<MM * CC / 256, 256, 0, stream>>>(ybuf, offbuf, maskbuf, dcn_bf);

    // MLP: fc1 (GELU, bf16 out) then fc2 (NCHW f32 out)
    wmma_gemm_kernel<2, 4><<<MM / 128, 256, 0, stream>>>(
        dcn_bf, wb_fc1, bp_fc1, hdn_bf, MM, HID);
    wmma_gemm_kernel<3, 8><<<MM / 128, 256, 0, stream>>>(
        hdn_bf, wb_fc2, bp_fc2, d_out, MM, CC);

    (void)in_sizes; (void)n_in; (void)out_size; (void)ws_size;
}
